// spk_vq_vae_lstm_50294067036784
// MI455X (gfx1250) — matrix operand order repaired and checked by reading, not compiled, not run on hardware
//
#include <hip/hip_runtime.h>
#include <hip/hip_bf16.h>

typedef __attribute__((ext_vector_type(16))) __bf16 v16bf;
typedef __attribute__((ext_vector_type(8)))  float  v8f;

#define KCODES 512
#define DDIM   64
#define ROWS_PER_BLOCK 128

// ---------------- zero workspace ----------------
__global__ void vq_zero(float* __restrict__ p, int n) {
  int i = blockIdx.x * blockDim.x + threadIdx.x;
  if (i < n) p[i] = 0.0f;
}

// ---------------- phase 1: argmin via split-bf16 WMMA + gather + scatter-add ----------------
__global__ __launch_bounds__(256) void vq_phase1(
    const float* __restrict__ x, const float* __restrict__ w,
    float* __restrict__ zq, float* __restrict__ sums, float* __restrict__ counts)
{
  extern __shared__ char smem[];
  __bf16* w_hi = (__bf16*)smem;                       // 512*64*2 = 65536 B
  __bf16* w_lo = (__bf16*)(smem + 65536);             // 65536 B
  float*  w2   = (float*) (smem + 131072);            // 2048 B

  const int  t         = threadIdx.x;
  const long blockRow0 = (long)blockIdx.x * ROWS_PER_BLOCK;

  // --- cooperative: split codebook into bf16 hi/lo in LDS; compute ||w_k||^2 ---
  #pragma unroll
  for (int r = 0; r < 2; ++r) {
    const int code = t * 2 + r;                        // 0..511
    const float* wr = w + code * DDIM;
    float acc = 0.0f;
    #pragma unroll
    for (int c = 0; c < DDIM; c += 4) {
      float4 v = *(const float4*)(wr + c);
      acc += v.x * v.x + v.y * v.y + v.z * v.z + v.w * v.w;
      __bf16 h0 = (__bf16)v.x, h1 = (__bf16)v.y, h2 = (__bf16)v.z, h3 = (__bf16)v.w;
      w_hi[code * DDIM + c + 0] = h0;
      w_hi[code * DDIM + c + 1] = h1;
      w_hi[code * DDIM + c + 2] = h2;
      w_hi[code * DDIM + c + 3] = h3;
      w_lo[code * DDIM + c + 0] = (__bf16)(v.x - (float)h0);
      w_lo[code * DDIM + c + 1] = (__bf16)(v.y - (float)h1);
      w_lo[code * DDIM + c + 2] = (__bf16)(v.z - (float)h2);
      w_lo[code * DDIM + c + 3] = (__bf16)(v.w - (float)h3);
    }
    w2[code] = acc;
  }
  __syncthreads();

  const int lane = t & 31;        // wave32
  const int wv   = t >> 5;        // 8 waves, 16 rows each
  const int half = lane >> 4;     // 0: low 16 lanes, 1: high 16 lanes
  const int m    = lane & 15;

  // --- A fragments (16x32 bf16, contiguous k order, same as B):
  //     half 0: K = {koff+0..15}; half 1: K = {koff+16..31}
  const long  arow = blockRow0 + wv * 16 + m;
  const float* xrow = x + arow * DDIM;
  v16bf a_hi[2], a_lo[2];
  #pragma unroll
  for (int f = 0; f < 2; ++f) {
    const int koff = f * 32;
    const float* p1 = xrow + koff + half * 16;
    const float* p2 = xrow + koff + 8 + half * 16;
    #pragma unroll
    for (int u = 0; u < 8; ++u) {
      float v1 = p1[u], v2 = p2[u];
      __bf16 h1 = (__bf16)v1, h2 = (__bf16)v2;
      a_hi[f][u]     = h1;
      a_hi[f][8 + u] = h2;
      a_lo[f][u]     = (__bf16)(v1 - (float)h1);
      a_lo[f][8 + u] = (__bf16)(v2 - (float)h2);
    }
  }

  float minv[8];
  int   mini[8];
  #pragma unroll
  for (int j = 0; j < 8; ++j) { minv[j] = 3.4e38f; mini[j] = 0; }

  // --- sweep all 512 codes in 16-wide tiles; 6 WMMAs/tile (bf16x3 compensation).
  //     Double-buffered B fragments (prefetch tile t+1 during tile t) and two
  //     independent accumulator chains (c0: K=0..31 terms, c1: K=32..63 terms).
  v16bf bh0[2], bh1[2], bl0[2], bl1[2];
  float w2v[2];

  #define LOAD_B(buf, tl)                                           \
    {                                                               \
      const int code_ = (tl) * 16 + m;                              \
      const __bf16* bh_ = w_hi + code_ * DDIM + half * 16;          \
      const __bf16* bl_ = w_lo + code_ * DDIM + half * 16;          \
      bh0[buf] = *(const v16bf*)(bh_);                              \
      bh1[buf] = *(const v16bf*)(bh_ + 32);                         \
      bl0[buf] = *(const v16bf*)(bl_);                              \
      bl1[buf] = *(const v16bf*)(bl_ + 32);                         \
      w2v[buf] = w2[code_];                                         \
    }

  LOAD_B(0, 0)
  #pragma unroll 2
  for (int tile = 0; tile < KCODES / 16; ++tile) {
    const int cur = tile & 1;
    const int nxt = cur ^ 1;
    LOAD_B(nxt, (tile + 1) & 31)          // prefetch next tile's B + w2

    v8f c0 = {}, c1 = {};
    c0 = __builtin_amdgcn_wmma_f32_16x16x32_bf16(false, a_hi[0], false, bh0[cur], (short)0, c0, false, false);
    c1 = __builtin_amdgcn_wmma_f32_16x16x32_bf16(false, a_hi[1], false, bh1[cur], (short)0, c1, false, false);
    c0 = __builtin_amdgcn_wmma_f32_16x16x32_bf16(false, a_lo[0], false, bh0[cur], (short)0, c0, false, false);
    c1 = __builtin_amdgcn_wmma_f32_16x16x32_bf16(false, a_lo[1], false, bh1[cur], (short)0, c1, false, false);
    c0 = __builtin_amdgcn_wmma_f32_16x16x32_bf16(false, a_hi[0], false, bl0[cur], (short)0, c0, false, false);
    c1 = __builtin_amdgcn_wmma_f32_16x16x32_bf16(false, a_hi[1], false, bl1[cur], (short)0, c1, false, false);

    const float w2c  = w2v[cur];
    const int   code = tile * 16 + m;     // C slot j: row = j + half*8, col = m
    #pragma unroll
    for (int j = 0; j < 8; ++j) {
      float dot  = c0[j] + c1[j];
      float dist = w2c - 2.0f * dot;
      if (dist < minv[j]) { minv[j] = dist; mini[j] = code; }
    }
  }
  #undef LOAD_B

  // --- argmin reduce across the 16 lanes of each half (masks <=8 stay in-half) ---
  #pragma unroll
  for (int off = 8; off; off >>= 1) {
    #pragma unroll
    for (int j = 0; j < 8; ++j) {
      float ov = __shfl_xor(minv[j], off, 32);
      int   oi = __shfl_xor(mini[j], off, 32);
      if (ov < minv[j] || (ov == minv[j] && oi < mini[j])) { minv[j] = ov; mini[j] = oi; }
    }
  }

  // --- emit z_q rows (gather from L2-hot f32 codebook) + scatter-add for segment mean ---
  #pragma unroll
  for (int j = 0; j < 8; ++j) {
    const long gr  = blockRow0 + wv * 16 + j + half * 8;
    const int  idx = mini[j];
    float4 wv4 = *(const float4*)(w + (size_t)idx * DDIM + m * 4);
    *(float4*)(zq + gr * DDIM + m * 4) = wv4;
    float4 xv = *(const float4*)(x + gr * DDIM + m * 4);   // L2-hot re-read
    atomicAdd(sums + idx * DDIM + m * 4 + 0, xv.x);
    atomicAdd(sums + idx * DDIM + m * 4 + 1, xv.y);
    atomicAdd(sums + idx * DDIM + m * 4 + 2, xv.z);
    atomicAdd(sums + idx * DDIM + m * 4 + 3, xv.w);
    if (m == 0) atomicAdd(counts + idx, 1.0f);
  }
}

// ---------------- phase 2: dict_grad = -2 * (mean_or_w - w) ----------------
__global__ void vq_finalize(const float* __restrict__ w, const float* __restrict__ sums,
                            const float* __restrict__ counts, float* __restrict__ grad)
{
  int idx = blockIdx.x * blockDim.x + threadIdx.x;
  if (idx >= KCODES * DDIM) return;
  int k = idx >> 6;
  float c = counts[k];
  float g = 0.0f;
  if (c > 0.0f) g = -2.0f * (sums[idx] / c - w[idx]);
  grad[idx] = g;
}

extern "C" void kernel_launch(void* const* d_in, const int* in_sizes, int n_in,
                              void* d_out, int out_size, void* d_ws, size_t ws_size,
                              hipStream_t stream) {
  (void)n_in; (void)out_size; (void)ws_size;
  const float* x = (const float*)d_in[0];
  const float* w = (const float*)d_in[1];
  const long N = (long)in_sizes[0] / DDIM;

  float* zq     = (float*)d_out;
  float* grad   = (float*)d_out + (size_t)N * DDIM;
  float* sums   = (float*)d_ws;              // [512*64]
  float* counts = sums + KCODES * DDIM;      // [512]

  const int zn = KCODES * DDIM + KCODES;
  vq_zero<<<(zn + 255) / 256, 256, 0, stream>>>(sums, zn);

  const int nblocks = (int)(N / ROWS_PER_BLOCK);     // N = 262144 -> 2048
  const size_t shmem = 65536 + 65536 + 2048;         // wh + wl + w2 = 130 KB/WGP-block
  vq_phase1<<<nblocks, 256, shmem, stream>>>(x, w, zq, sums, counts);

  vq_finalize<<<(KCODES * DDIM + 255) / 256, 256, 0, stream>>>(w, sums, counts, grad);
}
